// MixedAttentionLayer_40613210751286
// MI455X (gfx1250) — matrix operand order repaired and checked by reading, not compiled, not run on hardware
//
#include <hip/hip_runtime.h>
#include <math.h>

#define NNODE 6144
#define NEDGE 1024
#define NDIM  128
#define SCL   0.08838834764831845f   // 1/sqrt(128)

typedef __attribute__((ext_vector_type(8)))  float  v8f;
typedef __attribute__((ext_vector_type(2)))  float  v2f;
typedef __attribute__((ext_vector_type(16))) __bf16 v16bf;

__device__ __forceinline__ unsigned short f2bf(float f) {
  unsigned u = __float_as_uint(f);
  unsigned r = 0x7FFFu + ((u >> 16) & 1u);     // round-to-nearest-even
  return (unsigned short)((u + r) >> 16);
}

union BfFrag { v16bf v; unsigned u[8]; unsigned short s[16]; };

// ---- CDNA5 async global->LDS copy (ASYNCcnt protocol, ISA 08 §4) ----------
__device__ __forceinline__ void async_copy_b128(unsigned lds_off, unsigned long long gaddr) {
  asm volatile("global_load_async_to_lds_b128 %0, %1, off"
               :: "v"(lds_off), "v"(gaddr) : "memory");
}
__device__ __forceinline__ void wait_async() {
  asm volatile("s_wait_asynccnt 0x0" ::: "memory");
}

// ---------------------------------------------------------------------------
// bf16 WMMA GEMM, LDS block-tiled:
//   C[M,Nc] (+)= alpha * A[M,K] @ (TRANSB ? B[Nc,K]^T : B[K,Nc])
// 256 threads = 8 waves; block tile 128x128, BK=32; each wave does a 32x64
// region (2x4 tiles of 16x16). A (and B when TRANSB) staged into LDS with
// global_load_async_to_lds_b128; non-trans B transposed through LDS so the
// WMMA B fragment is always two contiguous ds_load_b128 per tile.
// Fragment layouts per CDNA5 ISA 7.12.2.
// ---------------------------------------------------------------------------
#define LDSROW 40   // 32 bf16 data + 8 pad; 80B row stride keeps 16B alignment
template<bool TRANSB, bool ACCUM>
__global__ __launch_bounds__(256)
void gemm_bf16_tiled(const unsigned short* __restrict__ A,
                     const unsigned short* __restrict__ B,
                     float* __restrict__ C,
                     int M, int Nc, int K, float alpha)
{
  __shared__ __align__(16) unsigned short As[128 * LDSROW];
  __shared__ __align__(16) unsigned short Bs[128 * LDSROW];

  const int tid  = threadIdx.x;
  const int lane = tid & 31;
  const int wave = tid >> 5;
  const int h    = lane >> 4;       // lane half
  const int l16  = lane & 15;
  const int wr   = wave >> 1;       // wave row 0..3  (32 rows each)
  const int wc   = wave & 1;        // wave col 0..1  (64 cols each)
  const int m0b  = blockIdx.y << 7;
  const int n0b  = blockIdx.x << 7;

  v8f acc[2][4] = {};

  for (int kk = 0; kk < K; kk += 32) {
    // ---- stage A tile: 128 rows x 32 bf16, 512 x 16B async transfers ----
#pragma unroll
    for (int it = 0; it < 2; ++it) {
      const int idx = tid + it * 256;         // 0..511
      const int row = idx >> 2, seg = idx & 3;
      const unsigned loff = (unsigned)(size_t)&As[row * LDSROW + seg * 8];
      const unsigned short* g = A + (size_t)(m0b + row) * K + kk + seg * 8;
      async_copy_b128(loff, (unsigned long long)(size_t)g);
    }
    if (TRANSB) {
      // B[Nc,K]: rows are contiguous in K -> same async pattern
#pragma unroll
      for (int it = 0; it < 2; ++it) {
        const int idx = tid + it * 256;
        const int row = idx >> 2, seg = idx & 3;
        const unsigned loff = (unsigned)(size_t)&Bs[row * LDSROW + seg * 8];
        const unsigned short* g = B + (size_t)(n0b + row) * K + kk + seg * 8;
        async_copy_b128(loff, (unsigned long long)(size_t)g);
      }
    } else {
      // B[K,Nc]: coalesced 16B global loads, transpose on LDS store
#pragma unroll
      for (int it = 0; it < 2; ++it) {
        const int idx = tid + it * 256;       // 0..511
        const int k = idx >> 4, seg = idx & 15;
        const uint4 q = *(const uint4*)(B + (size_t)(kk + k) * Nc + n0b + seg * 8);
        const unsigned v[4] = {q.x, q.y, q.z, q.w};
#pragma unroll
        for (int t = 0; t < 4; ++t) {
          Bs[(seg * 8 + 2 * t)     * LDSROW + k] = (unsigned short)(v[t] & 0xFFFFu);
          Bs[(seg * 8 + 2 * t + 1) * LDSROW + k] = (unsigned short)(v[t] >> 16);
        }
      }
    }
    wait_async();
    __syncthreads();

    // ---- fragments from LDS (ds_load_b128 pairs) ----
    BfFrag afr[2], bfr[4];
#pragma unroll
    for (int tm = 0; tm < 2; ++tm) {
      const int row = wr * 32 + tm * 16 + l16;
      const uint4 q0 = *(const uint4*)&As[row * LDSROW + h * 8];        // K h*8..+7
      const uint4 q1 = *(const uint4*)&As[row * LDSROW + 16 + h * 8];   // K 16+h*8..
      afr[tm].u[0] = q0.x; afr[tm].u[1] = q0.y; afr[tm].u[2] = q0.z; afr[tm].u[3] = q0.w;
      afr[tm].u[4] = q1.x; afr[tm].u[5] = q1.y; afr[tm].u[6] = q1.z; afr[tm].u[7] = q1.w;
    }
#pragma unroll
    for (int tn = 0; tn < 4; ++tn) {
      const int col = wc * 64 + tn * 16 + l16;
      const uint4 q0 = *(const uint4*)&Bs[col * LDSROW + h * 8];        // K h*8..+7
      const uint4 q1 = *(const uint4*)&Bs[col * LDSROW + 16 + h * 8];   // K 16+h*8..
      bfr[tn].u[0] = q0.x; bfr[tn].u[1] = q0.y; bfr[tn].u[2] = q0.z; bfr[tn].u[3] = q0.w;
      bfr[tn].u[4] = q1.x; bfr[tn].u[5] = q1.y; bfr[tn].u[6] = q1.z; bfr[tn].u[7] = q1.w;
    }
#pragma unroll
    for (int tm = 0; tm < 2; ++tm)
#pragma unroll
      for (int tn = 0; tn < 4; ++tn)
        acc[tm][tn] = __builtin_amdgcn_wmma_f32_16x16x32_bf16(
            false, afr[tm].v, false, bfr[tn].v, (short)0, acc[tm][tn], false, false);
    __syncthreads();
  }

  // ---- store: C/D VGPR r -> row +r (+8 for lane half h) ----
#pragma unroll
  for (int tm = 0; tm < 2; ++tm)
#pragma unroll
    for (int tn = 0; tn < 4; ++tn) {
      float* crow = C + (size_t)(m0b + wr * 32 + tm * 16 + h * 8) * Nc
                      + n0b + wc * 64 + tn * 16 + l16;
#pragma unroll
      for (int r = 0; r < 8; ++r) {
        const size_t idx = (size_t)r * Nc;
        if (ACCUM) crow[idx] += alpha * acc[tm][tn][r];
        else       crow[idx]  = alpha * acc[tm][tn][r];
      }
    }
}

// ---------------------------------------------------------------------------
// f32 WMMA projection GEMM: C[M,128] = A[M,128] @ W[128,128] (row major)
// ---------------------------------------------------------------------------
__global__ void gemm_f32_proj(const float* __restrict__ A,
                              const float* __restrict__ W,
                              float* __restrict__ C, int M)
{
  const int lane = threadIdx.x;
  const int h    = lane >> 4;
  const int l16  = lane & 15;
  const int m0   = (blockIdx.y * blockDim.y + threadIdx.y) << 4;
  const int n0   = blockIdx.x << 4;
  const float* arow = A + (size_t)(m0 + l16) * NDIM;
#if __has_builtin(__builtin_amdgcn_wmma_f32_16x16x4_f32)
  v8f acc = {};
  for (int kk = 0; kk < NDIM; kk += 4) {
    v2f a, b;
    a.x = arow[kk + 2 * h];
    a.y = arow[kk + 2 * h + 1];
    const float* bp = W + (size_t)(kk + 2 * h) * NDIM + n0 + l16;
    b.x = bp[0];
    b.y = bp[NDIM];
    acc = __builtin_amdgcn_wmma_f32_16x16x4_f32(false, a, false, b,
                                                (short)0, acc, false, false);
  }
  float* crow = C + (size_t)(m0 + h * 8) * NDIM + n0 + l16;
#pragma unroll
  for (int r = 0; r < 8; ++r) crow[(size_t)r * NDIM] = acc[r];
#else
  const int col = n0 + l16;
  float accs[8];
#pragma unroll
  for (int r = 0; r < 8; ++r) {
    const float* ar = A + (size_t)(m0 + h * 8 + r) * NDIM;
    float s = 0.f;
    for (int k = 0; k < NDIM; ++k) s += ar[k] * W[(size_t)k * NDIM + col];
    accs[r] = s;
  }
  float* crow = C + (size_t)(m0 + h * 8) * NDIM + col;
#pragma unroll
  for (int r = 0; r < 8; ++r) crow[(size_t)r * NDIM] = accs[r];
#endif
}

// ---------------------------------------------------------------------------
// per-node dot products: s_node = (h_he@u)*scale ; s1 = h_g@a[:D]; s2 = h_g@a[D:]
// ---------------------------------------------------------------------------
__global__ void vec3_kernel(const float* __restrict__ h_he,
                            const float* __restrict__ h_g,
                            const float* __restrict__ u,
                            const float* __restrict__ a_attn,
                            float* __restrict__ s_node,
                            float* __restrict__ s1,
                            float* __restrict__ s2)
{
  const int i = blockIdx.x * blockDim.x + threadIdx.x;
  if (i >= NNODE) return;
  const float* r1 = h_he + (size_t)i * NDIM;
  const float* r2 = h_g  + (size_t)i * NDIM;
  float d0 = 0.f, d1 = 0.f, d2 = 0.f;
  for (int k = 0; k < NDIM; ++k) {
    d0 += r1[k] * u[k];
    d1 += r2[k] * a_attn[k];
    d2 += r2[k] * a_attn[NDIM + k];
  }
  s_node[i] = d0 * SCL;
  s1[i] = d1;
  s2[i] = d2;
}

__device__ __forceinline__ float blkMax(float v, float* red) {
  const int tid = threadIdx.x;
  red[tid] = v; __syncthreads();
  for (int s = 128; s > 0; s >>= 1) { if (tid < s) red[tid] = fmaxf(red[tid], red[tid + s]); __syncthreads(); }
  float r = red[0]; __syncthreads(); return r;
}
__device__ __forceinline__ float blkSum(float v, float* red) {
  const int tid = threadIdx.x;
  red[tid] = v; __syncthreads();
  for (int s = 128; s > 0; s >>= 1) { if (tid < s) red[tid] += red[tid + s]; __syncthreads(); }
  float r = red[0]; __syncthreads(); return r;
}

// ---------------------------------------------------------------------------
// A = softmax(scores, axis=0)*mH ; column (edge) softmax over nodes.
// Also emits A16, At16 (transposed), mH16, he_sizes (clipped >= 1).
// ---------------------------------------------------------------------------
__global__ void colsoftmax_A(const float* __restrict__ H,
                             const float* __restrict__ s_node,
                             float* __restrict__ A,
                             unsigned short* __restrict__ A16,
                             unsigned short* __restrict__ At16,
                             unsigned short* __restrict__ mH16,
                             float* __restrict__ he_sizes)
{
  __shared__ float red[256];
  const int e = blockIdx.x, tid = threadIdx.x;
  float mx = -3.4e38f, cnt = 0.f;
  for (int i = tid; i < NNODE; i += 256)
    if (H[(size_t)i * NEDGE + e] > 0.f) { mx = fmaxf(mx, s_node[i]); cnt += 1.f; }
  mx  = blkMax(mx, red);
  cnt = blkSum(cnt, red);
  float sum = 0.f;
  for (int i = tid; i < NNODE; i += 256)
    if (H[(size_t)i * NEDGE + e] > 0.f) sum += __expf(s_node[i] - mx);
  sum = blkSum(sum, red);
  const float inv = (sum > 0.f) ? 1.f / sum : 0.f;
  for (int i = tid; i < NNODE; i += 256) {
    const bool m = H[(size_t)i * NEDGE + e] > 0.f;
    const float a = m ? __expf(s_node[i] - mx) * inv : 0.f;
    A[(size_t)i * NEDGE + e] = a;
    const unsigned short ab = f2bf(a);
    A16[(size_t)i * NEDGE + e] = ab;
    At16[(size_t)e * NNODE + i] = ab;
    mH16[(size_t)i * NEDGE + e] = m ? (unsigned short)0x3F80u : (unsigned short)0u;
  }
  if (tid == 0) he_sizes[e] = fmaxf(cnt, 1.f);
}

// ---------------------------------------------------------------------------
// masked row softmax over edges: X[n,:] = softmax(mask? X+Madd : NEG)*mask
// writes X (f32, in place) and X16 (bf16). Madd may be null.
// ---------------------------------------------------------------------------
__global__ void rowsoftmax_maskH(float* __restrict__ X,
                                 const float* __restrict__ H,
                                 unsigned short* __restrict__ X16,
                                 const float* __restrict__ Madd,
                                 int cols)
{
  __shared__ float red[256];
  const int n = blockIdx.x, tid = threadIdx.x;
  const size_t base = (size_t)n * cols;
  float mx = -3.4e38f;
  for (int c = tid; c < cols; c += 256)
    if (H[base + c] > 0.f) {
      float v = X[base + c] + (Madd ? Madd[base + c] : 0.f);
      mx = fmaxf(mx, v);
    }
  mx = blkMax(mx, red);
  float sum = 0.f;
  for (int c = tid; c < cols; c += 256)
    if (H[base + c] > 0.f) {
      float v = X[base + c] + (Madd ? Madd[base + c] : 0.f);
      sum += __expf(v - mx);
    }
  sum = blkSum(sum, red);
  const float inv = (sum > 0.f) ? 1.f / sum : 0.f;
  for (int c = tid; c < cols; c += 256) {
    const bool m = H[base + c] > 0.f;
    float o = 0.f;
    if (m) {
      float v = X[base + c] + (Madd ? Madd[base + c] : 0.f);
      o = __expf(v - mx) * inv;
    }
    X[base + c] = o;
    X16[base + c] = f2bf(o);
  }
}

// ---------------------------------------------------------------------------
// G = softmax over j of: lrelu(s1_i+s2_j,0.2)*f - 1e9*max(1-f,0), f = adj+eye
// ---------------------------------------------------------------------------
__device__ __forceinline__ float g_logit(const float* adj, const float* s2,
                                         int i, int j, float a) {
  const float f = (adj[(size_t)i * NNODE + j] > 0.f ? 1.f : 0.f) + (j == i ? 1.f : 0.f);
  const float x = a + s2[j];
  const float lr = x > 0.f ? x : 0.2f * x;
  return lr * f - 1e9f * fmaxf(1.f - f, 0.f);
}

__global__ void g_kernel(const float* __restrict__ adj,
                         const float* __restrict__ s1,
                         const float* __restrict__ s2,
                         float* __restrict__ S,
                         unsigned short* __restrict__ G16)
{
  __shared__ float red[256];
  const int i = blockIdx.x, tid = threadIdx.x;
  const float a = s1[i];
  float mx = -3.4e38f;
  for (int j = tid; j < NNODE; j += 256) mx = fmaxf(mx, g_logit(adj, s2, i, j, a));
  mx = blkMax(mx, red);
  float sum = 0.f;
  for (int j = tid; j < NNODE; j += 256) sum += __expf(g_logit(adj, s2, i, j, a) - mx);
  sum = blkSum(sum, red);
  const float inv = 1.f / sum;
  for (int j = tid; j < NNODE; j += 256) {
    const float g = __expf(g_logit(adj, s2, i, j, a) - mx) * inv;
    S[(size_t)i * NNODE + j] = g;
    G16[(size_t)i * NNODE + j] = f2bf(g);
  }
}

// A_hat[e,:] = softmax over n of (A[n,e]+M[n,e]) * mH[n,e]  -> bf16 [E,N]
__global__ void ahat_kernel(const float* __restrict__ A,
                            const float* __restrict__ Mb,
                            const float* __restrict__ H,
                            unsigned short* __restrict__ Ah16)
{
  __shared__ float red[256];
  const int e = blockIdx.x, tid = threadIdx.x;
  float mx = -3.4e38f;
  for (int n = tid; n < NNODE; n += 256)
    mx = fmaxf(mx, A[(size_t)n * NEDGE + e] + Mb[(size_t)n * NEDGE + e]);
  mx = blkMax(mx, red);
  float sum = 0.f;
  for (int n = tid; n < NNODE; n += 256)
    sum += __expf(A[(size_t)n * NEDGE + e] + Mb[(size_t)n * NEDGE + e] - mx);
  sum = blkSum(sum, red);
  const float inv = 1.f / sum;
  for (int n = tid; n < NNODE; n += 256) {
    const float m = (H[(size_t)n * NEDGE + e] > 0.f) ? 1.f : 0.f;
    const float v = __expf(A[(size_t)n * NEDGE + e] + Mb[(size_t)n * NEDGE + e] - mx) * inv * m;
    Ah16[(size_t)e * NNODE + n] = f2bf(v);
  }
}

// plain row softmax over N cols of S (in place) + bf16 copy
__global__ void rowsoftmax_plain(float* __restrict__ S,
                                 unsigned short* __restrict__ S16)
{
  __shared__ float red[256];
  const int i = blockIdx.x, tid = threadIdx.x;
  const size_t base = (size_t)i * NNODE;
  float mx = -3.4e38f;
  for (int j = tid; j < NNODE; j += 256) mx = fmaxf(mx, S[base + j]);
  mx = blkMax(mx, red);
  float sum = 0.f;
  for (int j = tid; j < NNODE; j += 256) sum += __expf(S[base + j] - mx);
  sum = blkSum(sum, red);
  const float inv = 1.f / sum;
  for (int j = tid; j < NNODE; j += 256) {
    const float g = __expf(S[base + j] - mx) * inv;
    S[base + j] = g;
    S16[base + j] = f2bf(g);
  }
}

__global__ void f2bf_kernel(const float* __restrict__ in,
                            unsigned short* __restrict__ out, int n) {
  const int i = blockIdx.x * blockDim.x + threadIdx.x;
  if (i < n) out[i] = f2bf(in[i]);
}

__global__ void div_hes_kernel(float* __restrict__ Mb,
                               const float* __restrict__ hes, int n) {
  const int i = blockIdx.x * blockDim.x + threadIdx.x;
  if (i < n) Mb[i] /= hes[i & (NEDGE - 1)];
}

__global__ void elu_kernel(float* __restrict__ x, int n) {
  const int i = blockIdx.x * blockDim.x + threadIdx.x;
  if (i < n) { const float v = x[i]; x[i] = v > 0.f ? v : __expf(v) - 1.f; }
}

// ---------------------------------------------------------------------------
extern "C" void kernel_launch(void* const* d_in, const int* in_sizes, int n_in,
                              void* d_out, int out_size, void* d_ws, size_t ws_size,
                              hipStream_t stream)
{
  const float* x    = (const float*)d_in[0];
  const float* H    = (const float*)d_in[1];
  const float* adj  = (const float*)d_in[2];
  const float* W_he = (const float*)d_in[3];
  const float* u    = (const float*)d_in[4];
  const float* W_e  = (const float*)d_in[5];
  const float* W_n  = (const float*)d_in[6];
  const float* W_g  = (const float*)d_in[7];
  const float* a_at = (const float*)d_in[8];

  float* out = (float*)d_out;
  float* zh = out;
  float* zg = out + (size_t)NNODE * NDIM;

  // workspace carve-up (fp32 first, then bf16; all sizes multiples of 8 ->
  // every buffer stays 16B-aligned for b128 access)
  float* fp = (float*)d_ws;
  auto allocF = [&](size_t n) { float* p = fp; fp += n; return p; };
  float* S    = allocF((size_t)NNODE * NNODE);   // G -> +BBt+AAt -> G_hat
  float* h_he = allocF((size_t)NNODE * NDIM);
  float* hn   = allocF((size_t)NNODE * NDIM);
  float* h_g  = allocF((size_t)NNODE * NDIM);
  float* Ab   = allocF((size_t)NNODE * NEDGE);
  float* Bb   = allocF((size_t)NNODE * NEDGE);
  float* Mb   = allocF((size_t)NNODE * NEDGE);
  float* erep = allocF((size_t)NEDGE * NDIM);
  float* hePr = allocF((size_t)NEDGE * NDIM);
  float* emix = allocF((size_t)NEDGE * NDIM);
  float* ewPr = allocF((size_t)NEDGE * NDIM);
  float* s_nd = allocF(NNODE);
  float* s1   = allocF(NNODE);
  float* s2   = allocF(NNODE);
  float* hes  = allocF(NEDGE);

  unsigned short* bp = (unsigned short*)fp;
  auto allocB = [&](size_t n) { unsigned short* p = bp; bp += n; return p; };
  unsigned short* G16   = allocB((size_t)NNODE * NNODE);
  unsigned short* A16   = allocB((size_t)NNODE * NEDGE);
  unsigned short* At16  = allocB((size_t)NEDGE * NNODE);
  unsigned short* B16   = allocB((size_t)NNODE * NEDGE);
  unsigned short* mH16  = allocB((size_t)NNODE * NEDGE);
  unsigned short* Ah16  = allocB((size_t)NEDGE * NNODE);
  unsigned short* Bh16  = allocB((size_t)NNODE * NEDGE);
  unsigned short* hhe16 = allocB((size_t)NNODE * NDIM);
  unsigned short* hn16  = allocB((size_t)NNODE * NDIM);
  unsigned short* hg16  = allocB((size_t)NNODE * NDIM);
  unsigned short* he16  = allocB((size_t)NEDGE * NDIM);
  unsigned short* ew16  = allocB((size_t)NEDGE * NDIM);

  const dim3 wblk(32, 4);               // projections: 4 waves, 1 tile each
  const int NF = NNODE * NDIM;
  const int EF = NEDGE * NDIM;

  // projections: h_he = x@W_he, hn = x@W_n, h_g = x@W_g  (f32 WMMA)
  gemm_f32_proj<<<dim3(NDIM / 16, NNODE / 64), wblk, 0, stream>>>(x, W_he, h_he, NNODE);
  gemm_f32_proj<<<dim3(NDIM / 16, NNODE / 64), wblk, 0, stream>>>(x, W_n,  hn,   NNODE);
  gemm_f32_proj<<<dim3(NDIM / 16, NNODE / 64), wblk, 0, stream>>>(x, W_g,  h_g,  NNODE);
  // per-node scalars
  vec3_kernel<<<NNODE / 256, 256, 0, stream>>>(h_he, h_g, u, a_at, s_nd, s1, s2);
  // A column softmax (+ bf16 copies, mask, he_sizes)
  colsoftmax_A<<<NEDGE, 256, 0, stream>>>(H, s_nd, Ab, A16, At16, mH16, hes);
  // bf16 copies of projections
  f2bf_kernel<<<(NF + 255) / 256, 256, 0, stream>>>(h_he, hhe16, NF);
  f2bf_kernel<<<(NF + 255) / 256, 256, 0, stream>>>(hn,   hn16,  NF);
  f2bf_kernel<<<(NF + 255) / 256, 256, 0, stream>>>(h_g,  hg16,  NF);
  // e_repr[E,D] = A^T @ h_he   (K = N)
  gemm_bf16_tiled<false, false><<<dim3(NDIM / 128, NEDGE / 128), 256, 0, stream>>>(
      At16, hhe16, erep, NEDGE, NDIM, NNODE, 1.f);
  // he = e_repr @ W_e
  gemm_f32_proj<<<dim3(NDIM / 16, NEDGE / 64), wblk, 0, stream>>>(erep, W_e, hePr, NEDGE);
  f2bf_kernel<<<(EF + 255) / 256, 256, 0, stream>>>(hePr, he16, EF);
  // B logits = scale * hn @ he^T   [N,E], K = D
  gemm_bf16_tiled<true, false><<<dim3(NEDGE / 128, NNODE / 128), 256, 0, stream>>>(
      hn16, he16, Bb, NNODE, NEDGE, NDIM, SCL);
  // B = masked row softmax
  rowsoftmax_maskH<<<NNODE, 256, 0, stream>>>(Bb, H, B16, (const float*)nullptr, NEDGE);
  // G row softmax -> S, G16
  g_kernel<<<NNODE, 256, 0, stream>>>(adj, s1, s2, S, G16);
  // M = G @ mH   [N,E], K = N ; then /= he_sizes
  gemm_bf16_tiled<false, false><<<dim3(NEDGE / 128, NNODE / 128), 256, 0, stream>>>(
      G16, mH16, Mb, NNODE, NEDGE, NNODE, 1.f);
  div_hes_kernel<<<(NNODE * NEDGE + 255) / 256, 256, 0, stream>>>(Mb, hes, NNODE * NEDGE);
  // A_hat[e,:] = softmax(A^T + M^T) * mH^T  -> bf16 [E,N]
  ahat_kernel<<<NEDGE, 256, 0, stream>>>(Ab, Mb, H, Ah16);
  // e_mixed = A_hat @ h_he ; ew = e_mixed @ W_e
  gemm_bf16_tiled<false, false><<<dim3(NDIM / 128, NEDGE / 128), 256, 0, stream>>>(
      Ah16, hhe16, emix, NEDGE, NDIM, NNODE, 1.f);
  gemm_f32_proj<<<dim3(NDIM / 16, NEDGE / 64), wblk, 0, stream>>>(emix, W_e, ewPr, NEDGE);
  f2bf_kernel<<<(EF + 255) / 256, 256, 0, stream>>>(ewPr, ew16, EF);
  // B_hat = softmax(B + M)*mH  (B16 already captured; overwriting Bb f32 is fine)
  rowsoftmax_maskH<<<NNODE, 256, 0, stream>>>(Bb, H, Bh16, Mb, NEDGE);
  // z_h = elu(B_hat @ ew)   [N,D], K = E
  gemm_bf16_tiled<false, false><<<dim3(NDIM / 128, NNODE / 128), 256, 0, stream>>>(
      Bh16, ew16, zh, NNODE, NDIM, NEDGE, 1.f);
  elu_kernel<<<(NF + 255) / 256, 256, 0, stream>>>(zh, NF);
  // S += B@B^T ; S += A@A^T   (the two 77-GFLOP GEMMs, bf16 WMMA, transB)
  gemm_bf16_tiled<true, true><<<dim3(NNODE / 128, NNODE / 128), 256, 0, stream>>>(
      B16, B16, S, NNODE, NNODE, NEDGE, 1.f);
  gemm_bf16_tiled<true, true><<<dim3(NNODE / 128, NNODE / 128), 256, 0, stream>>>(
      A16, A16, S, NNODE, NNODE, NEDGE, 1.f);
  // G_hat = row softmax(S) -> S, G16 (reuse)
  rowsoftmax_plain<<<NNODE, 256, 0, stream>>>(S, G16);
  // z_g = elu(G_hat @ h_g)   [N,D], K = N
  gemm_bf16_tiled<false, false><<<dim3(NDIM / 128, NNODE / 128), 256, 0, stream>>>(
      G16, hg16, zg, NNODE, NDIM, NNODE, 1.f);
  elu_kernel<<<(NF + 255) / 256, 256, 0, stream>>>(zg, NF);
}
